// DeepESN_56092272885773
// MI455X (gfx1250) — compile-verified, matrix operand order checked by reading, not run on hardware
//
#include <hip/hip_runtime.h>
#include <math.h>

// ---------------------------------------------------------------------------
// DeepESN on MI455X (gfx1250, wave32, WMMA).
//
// B=256, T=1024, D_IN=64, H=1024, L=3, alpha=0.5.
// pre = cur @ W_in^T + state @ W_res^T ; h = 0.5*state + 0.5*tanh(pre)
//
//  * f16 WMMA (v_wmma_f32_16x16x32_f16), fp32 accumulate. f16 weights
//    (~10.4 MB) live in the 192 MB L2 -> compute/latency bound.
//  * Layer-pipelined launches: launch p runs layer l at t = p - l; T+2
//    launches, graph-captured. Parity double-buffered state buffers.
//  * LDS tiles pre-swizzled to the ISA 16-bit fragment layout; staging is
//    GLOBAL_LOAD_ASYNC_TO_LDS_B128 (ASYNCcnt) into ping-pong LDS buffers,
//    one s_barrier per K-step, stage(s+1) overlapped with WMMA(s).
// ---------------------------------------------------------------------------

#define B_   256
#define T_   1024
#define DIN  64
#define H_   1024
#define L_   3
#define ALPHA_ 0.5f

#define MT 64    // M (batch) tile per workgroup
#define NT 128   // N (hidden) tile per workgroup
#define KT 32    // K step (WMMA K for f16)

#define LDSA_U (MT * KT / 2)   // uints per A buffer (1024 = 4 KB)
#define LDSB_U (NT * KT / 2)   // uints per B buffer (2048 = 8 KB)

typedef _Float16 v16h __attribute__((ext_vector_type(16)));
typedef float    v8f  __attribute__((ext_vector_type(8)));

#if __has_builtin(__builtin_amdgcn_global_load_async_to_lds_b128)
#define ASYNC_COPY 1
#else
#define ASYNC_COPY 0
#endif

#if ASYNC_COPY
typedef int vs4i __attribute__((vector_size(4 * sizeof(int))));
typedef __attribute__((address_space(1))) vs4i* as1_v4i;   // global v4i*
typedef __attribute__((address_space(3))) vs4i* as3_v4i;   // LDS v4i*
#endif

// ---- async (or fallback) 16B global -> LDS copy ----------------------------
__device__ __forceinline__ void cp16_g2l(const _Float16* __restrict__ g,
                                         unsigned* __restrict__ l) {
#if ASYNC_COPY
  __builtin_amdgcn_global_load_async_to_lds_b128(
      (as1_v4i)(void*)const_cast<_Float16*>(g),
      (as3_v4i)(void*)l, 0, 0);
#else
  *(uint4*)l = *(const uint4*)g;
#endif
}

__device__ __forceinline__ void wait_async() {
#if ASYNC_COPY
#if __has_builtin(__builtin_amdgcn_s_wait_asynccnt)
  __builtin_amdgcn_s_wait_asynccnt(0);
#else
  asm volatile("s_wait_asynccnt 0x0" ::: "memory");
#endif
#endif
}

// Fragment swizzle (ISA 7.12.2, 16-bit A/B operands):
//   lane = (row&15) | (((k>>3)&1)<<4);  element e = (k&7) | ((k>>4)<<3)
// Frag-major LDS tile: frag f = 256 uints (32 lanes x 8 f16-pairs).
// Quarter q (k = q*8..q*8+7) of a row -> 4 contiguous pairs: one 16B chunk.
__device__ __forceinline__ int chunk_idx(int row, int q) {
  int lane = (row & 15) | ((q & 1) << 4);
  return (row >> 4) * 256 + lane * 8 + (q >> 1) * 4;   // uint units, 16B-aligned
}

// ---- stage one NT x KT weight tile (W row-major [N][ldw]) ------------------
__device__ __forceinline__ void stage_B(const _Float16* __restrict__ W,
                                        int ldw, int kOff, int nBase,
                                        unsigned* __restrict__ ldsB, int tid) {
#pragma unroll
  for (int it = 0; it < (NT * KT / 8) / 256; ++it) {   // 2 chunks / thread
    int c = tid + it * 256;
    int n = c >> 2, q = c & 3;
    cp16_g2l(W + (size_t)(nBase + n) * ldw + kOff + q * 8,
             &ldsB[chunk_idx(n, q)]);
  }
}

// ---- stage one MT x KT activation tile from an f16 row-major matrix --------
__device__ __forceinline__ void stage_A(const _Float16* __restrict__ src0,
                                        int lda, int kOff, int mBase,
                                        unsigned* __restrict__ ldsA, int tid) {
  int m = tid >> 2, q = tid & 3;                       // 1 chunk / thread
  cp16_g2l(src0 + (size_t)(mBase + m) * lda + kOff + q * 8,
           &ldsA[chunk_idx(m, q)]);
}

// ---- stage one MT x KT tile from fp32 x[:, t, :], converting to f16 --------
// (synchronous: needs a conversion through VGPRs; only 2 steps per launch)
__device__ __forceinline__ void stage_A_x(const float* __restrict__ x, int t,
                                          int kOff, int mBase,
                                          unsigned* __restrict__ ldsA, int tid) {
  int m = tid >> 2, q = tid & 3;
  const float* sp = x + ((size_t)(mBase + m) * T_ + t) * DIN + kOff + q * 8;
  float4 f0 = *(const float4*)sp;
  float4 f1 = *(const float4*)(sp + 4);
  union { _Float16 h[8]; uint4 u; } cv;
  cv.h[0] = (_Float16)f0.x; cv.h[1] = (_Float16)f0.y;
  cv.h[2] = (_Float16)f0.z; cv.h[3] = (_Float16)f0.w;
  cv.h[4] = (_Float16)f1.x; cv.h[5] = (_Float16)f1.y;
  cv.h[6] = (_Float16)f1.z; cv.h[7] = (_Float16)f1.w;
  *(uint4*)&ldsA[chunk_idx(m, q)] = cv.u;
}

// ---- fragments from LDS + 4 WMMAs ------------------------------------------
__device__ __forceinline__ void mma_step(const unsigned* __restrict__ ldsA,
                                         const unsigned* __restrict__ ldsB,
                                         int wm, int wn, int lane, v8f acc[2][2]) {
  const v16h* af = (const v16h*)ldsA;
  const v16h* bf = (const v16h*)ldsB;
  v16h a0 = af[(wm * 2 + 0) * 32 + lane];
  v16h a1 = af[(wm * 2 + 1) * 32 + lane];
  v16h b0 = bf[(wn * 2 + 0) * 32 + lane];
  v16h b1 = bf[(wn * 2 + 1) * 32 + lane];
  acc[0][0] = __builtin_amdgcn_wmma_f32_16x16x32_f16(false, a0, false, b0,
                  (short)0, acc[0][0], false, false);
  acc[0][1] = __builtin_amdgcn_wmma_f32_16x16x32_f16(false, a0, false, b1,
                  (short)0, acc[0][1], false, false);
  acc[1][0] = __builtin_amdgcn_wmma_f32_16x16x32_f16(false, a1, false, b0,
                  (short)0, acc[1][0], false, false);
  acc[1][1] = __builtin_amdgcn_wmma_f32_16x16x32_f16(false, a1, false, b1,
                  (short)0, acc[1][1], false, false);
}

// ---- one GEMM phase, double-buffered, one barrier per K-step ---------------
// Safety: the barrier at step s+1 is reached only after every wave's WMMAs of
// step s executed, which forces their dscnt waits -> all reads of buffer
// (s&1) are complete before anyone overwrites it at step s+2.
template <bool XCONV>
__device__ __forceinline__ void gemm_phase(const float* __restrict__ x, int t,
                                           const _Float16* __restrict__ Asrc, int lda,
                                           const _Float16* __restrict__ Wsrc, int ldw,
                                           int steps,   // even
                                           int mBase, int nBase,
                                           int tid, int wm, int wn, int lane,
                                           unsigned (*__restrict__ ldsA)[LDSA_U],
                                           unsigned (*__restrict__ ldsB)[LDSB_U],
                                           v8f acc[2][2]) {
  if (XCONV) stage_A_x(x, t, 0, mBase, ldsA[0], tid);
  else       stage_A(Asrc, lda, 0, mBase, ldsA[0], tid);
  stage_B(Wsrc, ldw, 0, nBase, ldsB[0], tid);

  for (int s = 0; s < steps; s += 2) {
    wait_async();
    __syncthreads();
    if (s + 1 < steps) {
      if (XCONV) stage_A_x(x, t, (s + 1) * KT, mBase, ldsA[1], tid);
      else       stage_A(Asrc, lda, (s + 1) * KT, mBase, ldsA[1], tid);
      stage_B(Wsrc, ldw, (s + 1) * KT, nBase, ldsB[1], tid);
    }
    mma_step(ldsA[0], ldsB[0], wm, wn, lane, acc);

    wait_async();
    __syncthreads();
    if (s + 2 < steps) {
      if (XCONV) stage_A_x(x, t, (s + 2) * KT, mBase, ldsA[0], tid);
      else       stage_A(Asrc, lda, (s + 2) * KT, mBase, ldsA[0], tid);
      stage_B(Wsrc, ldw, (s + 2) * KT, nBase, ldsB[0], tid);
    }
    mma_step(ldsA[1], ldsB[1], wm, wn, lane, acc);
  }
}

// ---------------------------------------------------------------------------
__global__ void __launch_bounds__(256)
esn_step_kernel(const float* __restrict__ x,
                const _Float16* __restrict__ wIn0,   // [H][DIN] f16
                const _Float16* __restrict__ wInR,   // [2][H][H] f16
                const _Float16* __restrict__ wRes,   // [3][H][H] f16
                float*    __restrict__ stF32,        // [2][L][B][H]
                _Float16* __restrict__ stF16,        // [2][L][B][H]
                _Float16* __restrict__ curBuf,       // [2 layers][2 par][B][H]
                int p)
{
  const int l = blockIdx.z;
  const int t = p - l;                 // pipelined time index for this layer
  if (t < 0 || t >= T_) return;        // block-uniform

  const int tid   = threadIdx.x;
  const int lane  = tid & 31;
  const int wave  = tid >> 5;          // 8 waves
  const int wm    = wave >> 2;         // 0..1  (32 rows each)
  const int wn    = wave & 3;          // 0..3  (32 cols each)
  const int mBase = blockIdx.y * MT;
  const int nBase = blockIdx.x * NT;

  const _Float16* Wi = (l == 0) ? wIn0 : (wInR + (size_t)(l - 1) * H_ * H_);
  const _Float16* Wr = wRes + (size_t)l * H_ * H_;

  const int rp = p & 1, wp = rp ^ 1;   // read / write parity
  const float*    stR32 = stF32 + ((size_t)(rp * L_ + l) * B_) * H_;
  float*          stW32 = stF32 + ((size_t)(wp * L_ + l) * B_) * H_;
  const _Float16* stR16 = stF16 + ((size_t)(rp * L_ + l) * B_) * H_;
  _Float16*       stW16 = stF16 + ((size_t)(wp * L_ + l) * B_) * H_;
  const _Float16* curIn = (l == 0) ? nullptr
      : curBuf + ((size_t)((l - 1) * 2 + ((p - 1) & 1)) * B_) * H_;
  _Float16* curOut = (l < L_ - 1)
      ? curBuf + ((size_t)(l * 2 + (p & 1)) * B_) * H_ : nullptr;

  __shared__ __align__(32) unsigned ldsA[2][LDSA_U];  // 2 x 4 KB
  __shared__ __align__(32) unsigned ldsB[2][LDSB_U];  // 2 x 8 KB

  v8f acc[2][2] = {};

  // ---- phase 1: input contribution ---------------------------------------
  if (l == 0) {
    gemm_phase<true>(x, t, nullptr, 0, Wi, DIN, DIN / KT,     // 2 steps
                     mBase, nBase, tid, wm, wn, lane, ldsA, ldsB, acc);
  } else {
    gemm_phase<false>(nullptr, 0, curIn, H_, Wi, H_, H_ / KT, // 32 steps
                      mBase, nBase, tid, wm, wn, lane, ldsA, ldsB, acc);
  }

  // ---- phase 2: reservoir contribution ------------------------------------
  gemm_phase<false>(nullptr, 0, stR16, H_, Wr, H_, H_ / KT,   // 32 steps
                    mBase, nBase, tid, wm, wn, lane, ldsA, ldsB, acc);

  // ---- epilogue: leaky-tanh state update (fp32) ---------------------------
  // C/D layout: VGPR r -> M = r + 8*(lane>=16); N = lane & 15.
#pragma unroll
  for (int i = 0; i < 2; ++i) {
#pragma unroll
    for (int j = 0; j < 2; ++j) {
      int mtb = mBase + wm * 32 + i * 16 + ((lane >> 4) << 3);
      int n   = nBase + wn * 32 + j * 16 + (lane & 15);
#pragma unroll
      for (int r = 0; r < 8; ++r) {
        int m = mtb + r;
        size_t off = (size_t)m * H_ + n;
        float pre  = acc[i][j][r];
        float sOld = stR32[off];
        float h = (1.0f - ALPHA_) * sOld + ALPHA_ * tanhf(pre);
        stW32[off] = h;
        _Float16 hh = (_Float16)h;
        stW16[off] = hh;
        if (curOut) curOut[off] = hh;
      }
    }
  }
}

// ---------------------------------------------------------------------------
__global__ void cvt_f32_f16_kernel(const float* __restrict__ src,
                                   _Float16* __restrict__ dst, int n) {
  int i = blockIdx.x * 256 + threadIdx.x;
  if (i < n) dst[i] = (_Float16)src[i];
}

__global__ void zero_f32_kernel(float* p, int n) {
  int i = blockIdx.x * 256 + threadIdx.x;
  if (i < n) p[i] = 0.0f;
}

__global__ void zero_f16_kernel(_Float16* p, int n) {
  int i = blockIdx.x * 256 + threadIdx.x;
  if (i < n) p[i] = (_Float16)0.0f;
}

// out[b] = sum_{l,h} state[l][b][h] * w_out[l*H+h] + b_out, final parities
__global__ void __launch_bounds__(256)
readout_kernel(const float* __restrict__ stF32,
               const float* __restrict__ wout,
               const float* __restrict__ bout,
               float* __restrict__ out) {
  int b = blockIdx.x, tid = threadIdx.x;
  float acc = 0.0f;
  for (int idx = tid; idx < L_ * H_; idx += 256) {
    int l  = idx >> 10;
    int h2 = idx & (H_ - 1);
    int par = (T_ + l) & 1;   // layer l's last write (p = T-1+l) parity
    float v = stF32[(((size_t)par * L_ + l) * B_ + b) * H_ + h2];
    acc += v * wout[idx];
  }
  __shared__ float red[256];
  red[tid] = acc;
  __syncthreads();
  for (int s2 = 128; s2 > 0; s2 >>= 1) {
    if (tid < s2) red[tid] += red[tid + s2];
    __syncthreads();
  }
  if (tid == 0) out[b] = red[0] + bout[0];
}

// ---------------------------------------------------------------------------
extern "C" void kernel_launch(void* const* d_in, const int* in_sizes, int n_in,
                              void* d_out, int out_size, void* d_ws, size_t ws_size,
                              hipStream_t stream) {
  (void)in_sizes; (void)n_in; (void)out_size; (void)ws_size;

  const float* x    = (const float*)d_in[0];   // [B,T,DIN]
  const float* Win0 = (const float*)d_in[1];   // [H,DIN]
  const float* WinR = (const float*)d_in[2];   // [2,H,H]
  const float* Wres = (const float*)d_in[3];   // [3,H,H]
  const float* wout = (const float*)d_in[4];   // [1, L*H]
  const float* bout = (const float*)d_in[5];   // [1]
  float* out = (float*)d_out;

  // Workspace carve-out (~21.2 MB total)
  char* ws = (char*)d_ws;
  size_t off = 0;
  _Float16* wIn0h = (_Float16*)(ws + off); off += (size_t)H_ * DIN * 2;         // 128 KB
  _Float16* wInRh = (_Float16*)(ws + off); off += (size_t)2 * H_ * H_ * 2;      // 4 MB
  _Float16* wResh = (_Float16*)(ws + off); off += (size_t)3 * H_ * H_ * 2;      // 6 MB
  float*    stF32 = (float*)   (ws + off); off += (size_t)2 * L_ * B_ * H_ * 4; // 6 MB
  _Float16* stF16 = (_Float16*)(ws + off); off += (size_t)2 * L_ * B_ * H_ * 2; // 3 MB
  _Float16* curB  = (_Float16*)(ws + off); off += (size_t)2 * 2 * B_ * H_ * 2;  // 2 MB

  int n;
  n = H_ * DIN;
  cvt_f32_f16_kernel<<<(n + 255) / 256, 256, 0, stream>>>(Win0, wIn0h, n);
  n = 2 * H_ * H_;
  cvt_f32_f16_kernel<<<(n + 255) / 256, 256, 0, stream>>>(WinR, wInRh, n);
  n = 3 * H_ * H_;
  cvt_f32_f16_kernel<<<(n + 255) / 256, 256, 0, stream>>>(Wres, wResh, n);
  n = 2 * L_ * B_ * H_;
  zero_f32_kernel<<<(n + 255) / 256, 256, 0, stream>>>(stF32, n);
  zero_f16_kernel<<<(n + 255) / 256, 256, 0, stream>>>(stF16, n);

  // Pipelined recurrence: launch p runs layer l at time t = p - l.
  dim3 grid(H_ / NT, B_ / MT, L_);   // (8, 4, 3) = 96 workgroups / launch
  for (int p = 0; p < T_ + L_ - 1; ++p)
    esn_step_kernel<<<grid, 256, 0, stream>>>(x, wIn0h, wInRh, wResh,
                                              stF32, stF16, curB, p);

  readout_kernel<<<B_, 256, 0, stream>>>(stF32, wout, bout, out);
}
